// SRU_FedNP_46351287058786
// MI455X (gfx1250) — compile-verified, matrix operand order checked by reading, not run on hardware
//
#include <hip/hip_runtime.h>
#include <hip/hip_bf16.h>
#include <math.h>

typedef __attribute__((ext_vector_type(16))) __bf16 v16bf;
typedef __attribute__((ext_vector_type(8)))  float  v8f;
typedef __attribute__((ext_vector_type(2)))  __bf16 v2bf;
typedef __attribute__((ext_vector_type(2)))  float  v2f;

#define TILE_M 64
#define TILE_N 128
#define TILE_K 32
#define LDSS   40   // 32 + 8 pad (__bf16): rows 16B aligned, de-phased banks

// Split a float pair into packed bf16 hi/lo dwords using vector converts so the
// compiler emits packed v_cvt_pk_bf16_f32 (no sub-register repacking).
__device__ __forceinline__ void split2_bf16(float x0, float x1,
                                            unsigned int &hp, unsigned int &lp) {
    v2f x = {x0, x1};
    v2bf h = __builtin_convertvector(x, v2bf);          // packed RNE truncate
    v2f hf = __builtin_convertvector(h, v2f);
    v2bf l = __builtin_convertvector(x - hf, v2bf);     // packed residual
    hp = __builtin_bit_cast(unsigned int, h);
    lp = __builtin_bit_cast(unsigned int, l);
}

// C[M,N] = A[M,K] * B[K,N] (+bias), fp32 in/out, bf16x3 WMMA inside.
// Requires M % TILE_M == 0 and K even (true for all launches: M=2560, K in {200,1024}).
// relu=1: out_main = max(v,0); out_pre (if non-null) = v (pre-activation).
__global__ __launch_bounds__(256)
void gemm_bf16x3(const float* __restrict__ A, const float* __restrict__ Bm,
                 const float* __restrict__ bias,
                 float* __restrict__ out_main, float* __restrict__ out_pre,
                 int M, int N, int K, int relu)
{
    __shared__ __bf16 Ash[2][2][TILE_M][LDSS];   // [buf][hi/lo][row][k]
    __shared__ __bf16 Bsh[2][2][TILE_N][LDSS];   // [buf][hi/lo][col][k]

    const int tid  = threadIdx.x;
    const int lane = tid & 31;
    const int wave = tid >> 5;
    const int wm   = wave & 1;    // 2 waves along M
    const int wn   = wave >> 1;   // 4 waves along N
    const int m0   = blockIdx.y * TILE_M;
    const int n0   = blockIdx.x * TILE_N;

    // Pair-granular staging coordinates (K even => k-pairs never straddle K):
    // A tile 64x32 = 1024 pairs: thread -> (row = ar4 + 16i, kpair = akp), i = 0..3
    // B tile 32x128 = 2048 pairs: thread -> (kpair = bkp + 4i, col = bn), i = 0..7
    const int ar4 = tid >> 4;          // 0..15
    const int akp = (tid & 15) * 2;    // even k within chunk
    const int bkp = (tid >> 7) * 2;    // 0 or 2
    const int bn  = tid & 127;
    const int  gnB  = n0 + bn;
    const bool nOK  = gnB < N;
    const int  gnBc = nOK ? gnB : 0;

    v8f zero = {};
    v8f acc[2][2];
    acc[0][0] = zero; acc[0][1] = zero; acc[1][0] = zero; acc[1][1] = zero;

    const int lr = lane & 15;
    const int kb = (lane >> 4) * 8;   // half-wave K phase (ISA 16-bit A/B layout)

    const int nChunks = (K + TILE_K - 1) / TILE_K;

    float aR[8];    // 4 float2 pairs
    float bR[16];   // 8 pairs

    // Branch-free global loads (clamped offsets + select-zero, 32-bit offsets).
    auto load_regs = [&](int k0) {
        const int  gk   = k0 + akp;
        const bool akOK = gk < K;                       // pair-valid (K even)
        const float2* A2 = (const float2*)A;
        const int  ab2  = ((m0 + ar4) * K + (akOK ? gk : 0)) >> 1;
        #pragma unroll
        for (int i = 0; i < 4; ++i) {
            float2 v = A2[ab2 + i * 8 * K];             // +16 rows = 8*K float2
            aR[2*i]   = akOK ? v.x : 0.0f;
            aR[2*i+1] = akOK ? v.y : 0.0f;
        }
        #pragma unroll
        for (int i = 0; i < 8; ++i) {
            const int  k2  = k0 + bkp + 4 * i;
            const bool kOK = k2 < K;                    // pair-valid (K even)
            float v0 = Bm[(kOK ? k2     : 0) * N + gnBc];
            float v1 = Bm[(kOK ? k2 + 1 : 0) * N + gnBc];
            bR[2*i]   = (kOK && nOK) ? v0 : 0.0f;
            bR[2*i+1] = (kOK && nOK) ? v1 : 0.0f;
        }
    };
    auto store_lds = [&](int buf) {
        #pragma unroll
        for (int i = 0; i < 4; ++i) {
            unsigned int hp, lp;
            split2_bf16(aR[2*i], aR[2*i+1], hp, lp);
            *(unsigned int*)&Ash[buf][0][ar4 + i * 16][akp] = hp;
            *(unsigned int*)&Ash[buf][1][ar4 + i * 16][akp] = lp;
        }
        #pragma unroll
        for (int i = 0; i < 8; ++i) {
            unsigned int hp, lp;
            split2_bf16(bR[2*i], bR[2*i+1], hp, lp);
            *(unsigned int*)&Bsh[buf][0][bn][bkp + 4 * i] = hp;
            *(unsigned int*)&Bsh[buf][1][bn][bkp + 4 * i] = lp;
        }
    };

    // Prologue: stage chunk 0.
    load_regs(0);
    store_lds(0);
    __syncthreads();

    for (int c = 0; c < nChunks; ++c) {
        const int  buf     = c & 1;
        const bool hasNext = (c + 1) < nChunks;

        // Issue global loads for chunk c+1 now; they overlap this chunk's WMMAs.
        if (hasNext) {
            load_regs((c + 1) * TILE_K);
            if ((c + 2) < nChunks) {      // pull chunk c+2 toward L2
                const int pk = (c + 2) * TILE_K;
                __builtin_prefetch(&A[(m0 + ar4) * K + (pk + akp < K ? pk + akp : 0)], 0, 1);
                __builtin_prefetch(&Bm[((pk + bkp) < K ? (pk + bkp) : 0) * N + gnBc], 0, 1);
            }
        }

        // Fragments: elems 0..7 -> K = kb..kb+7; 8..15 -> K = 16+kb.. (b128 runs)
        v16bf ah[2], al[2], bh[2], bl[2];
        #pragma unroll
        for (int mt = 0; mt < 2; ++mt) {
            int row = wm * 32 + mt * 16 + lr;
            #pragma unroll
            for (int j = 0; j < 8; ++j) {
                ah[mt][j]     = Ash[buf][0][row][kb + j];
                ah[mt][j + 8] = Ash[buf][0][row][16 + kb + j];
                al[mt][j]     = Ash[buf][1][row][kb + j];
                al[mt][j + 8] = Ash[buf][1][row][16 + kb + j];
            }
        }
        #pragma unroll
        for (int nt = 0; nt < 2; ++nt) {
            int col = wn * 32 + nt * 16 + lr;
            #pragma unroll
            for (int j = 0; j < 8; ++j) {
                bh[nt][j]     = Bsh[buf][0][col][kb + j];
                bh[nt][j + 8] = Bsh[buf][0][col][16 + kb + j];
                bl[nt][j]     = Bsh[buf][1][col][kb + j];
                bl[nt][j + 8] = Bsh[buf][1][col][16 + kb + j];
            }
        }

        // bf16x3: hi*hi + hi*lo + lo*hi, f32 accumulate.
        #pragma unroll
        for (int mt = 0; mt < 2; ++mt)
            #pragma unroll
            for (int nt = 0; nt < 2; ++nt) {
                acc[mt][nt] = __builtin_amdgcn_wmma_f32_16x16x32_bf16(
                    false, ah[mt], false, bh[nt], (short)0, acc[mt][nt], false, false);
                acc[mt][nt] = __builtin_amdgcn_wmma_f32_16x16x32_bf16(
                    false, ah[mt], false, bl[nt], (short)0, acc[mt][nt], false, false);
                acc[mt][nt] = __builtin_amdgcn_wmma_f32_16x16x32_bf16(
                    false, al[mt], false, bh[nt], (short)0, acc[mt][nt], false, false);
            }

        if (hasNext) store_lds(buf ^ 1);   // other buffer: no hazard with reads above
        __syncthreads();
    }

    // Store: v8f elem e -> lanes 0-15: (M=e, N=lr); lanes 16-31: (M=e+8, N=lr)
    const int mh = (lane >> 4) * 8;
    #pragma unroll
    for (int mt = 0; mt < 2; ++mt)
        #pragma unroll
        for (int nt = 0; nt < 2; ++nt) {
            int col = n0 + wn * 32 + nt * 16 + lr;
            if (col >= N) continue;
            float bv = bias ? bias[col] : 0.0f;
            #pragma unroll
            for (int e = 0; e < 8; ++e) {
                int row = m0 + wm * 32 + mt * 16 + mh + e;
                float v = acc[mt][nt][e] + bv;
                int o = row * N + col;
                if (relu) {
                    if (out_pre) out_pre[o] = v;
                    out_main[o] = v > 0.0f ? v : 0.0f;
                } else {
                    out_main[o] = v;
                }
            }
        }
}

// SRU recurrence: parallel over (b,h), sequential over T.
__global__ __launch_bounds__(256)
void sru_scan(const float* __restrict__ U, const float* __restrict__ xin,
              const float* __restrict__ c0, const float* __restrict__ bf,
              const float* __restrict__ br,
              float* __restrict__ hout, float* __restrict__ cT,
              int T, int Bdim, int H)
{
    int idx = blockIdx.x * blockDim.x + threadIdx.x;
    if (idx >= Bdim * H) return;
    int h = idx % H;
    int b = idx / H;
    float c   = c0[b * H + h];
    float bfv = bf[h], brv = br[h];
    for (int t = 0; t < T; ++t) {
        int r = t * Bdim + b;
        const float* u = U + r * 3 * H;
        float xt = u[h];
        float f  = 1.0f / (1.0f + expf(-(u[H + h]   + bfv)));
        float rg = 1.0f / (1.0f + expf(-(u[2*H + h] + brv)));
        c = f * c + (1.0f - f) * xt;
        float hv = rg * tanhf(c) + (1.0f - rg) * xin[r * H + h];
        hout[r * H + h] = hv;
    }
    cT[b * H + h] = c;
}

// dst[D2,D1,H] <- src[D1,D2,H]  (swap first two dims)
__global__ __launch_bounds__(256)
void permute_swap01(float* __restrict__ dst, const float* __restrict__ src,
                    int D1, int D2, int H)
{
    int idx = blockIdx.x * blockDim.x + threadIdx.x;
    int total = D1 * D2 * H;
    if (idx >= total) return;
    int h  = idx % H;
    int ij = idx / H;
    int j = ij % D2;
    int i = ij / D2;
    dst[(j * D1 + i) * H + h] = src[idx];
}

extern "C" void kernel_launch(void* const* d_in, const int* in_sizes, int n_in,
                              void* d_out, int out_size, void* d_ws, size_t ws_size,
                              hipStream_t stream) {
    (void)in_sizes; (void)n_in; (void)out_size; (void)ws_size;
    const float* x      = (const float*)d_in[0];
    const float* hidden = (const float*)d_in[1];
    const float* W1     = (const float*)d_in[2];
    const float* b1     = (const float*)d_in[3];
    const float* sruW   = (const float*)d_in[4];
    const float* sruB   = (const float*)d_in[5];
    const float* W3     = (const float*)d_in[6];
    const float* b3     = (const float*)d_in[7];
    const float* W4     = (const float*)d_in[8];
    const float* b4     = (const float*)d_in[9];

    const int D_IN = 200, H = 1024, OUT = 1095, L = 12, B = 128, T = 20;
    const int M = B * T;  // 2560

    float* out    = (float*)d_out;
    float* logits = out;
    float* fx     = out + (long)M * OUT;
    float* hidA   = fx  + (long)M * H;

    float* ws   = (float*)d_ws;
    float* bufA = ws;                       // M*H
    float* bufB = bufA + (long)M * H;       // M*H
    float* bufU = bufB + (long)M * H;       // M*3H (reused as hbt at the end)

    dim3 blk(256);

    // Dense1: [M x 200] * [200 x 1024] + b1 -> bufB (row order b*T+t)
    gemm_bf16x3<<<dim3((H + TILE_N - 1) / TILE_N, M / TILE_M), blk, 0, stream>>>(
        x, W1, b1, bufB, nullptr, M, H, D_IN, 0);
    // [B,T,H] -> [T,B,H]
    permute_swap01<<<(M * H + 255) / 256, blk, 0, stream>>>(bufA, bufB, B, T, H);

    float* hin = bufA;
    float* hot = bufB;
    for (int l = 0; l < L; ++l) {
        gemm_bf16x3<<<dim3((3 * H + TILE_N - 1) / TILE_N, M / TILE_M), blk, 0, stream>>>(
            hin, sruW + (long)l * H * 3 * H, nullptr, bufU, nullptr, M, 3 * H, H, 0);
        sru_scan<<<(B * H + 255) / 256, blk, 0, stream>>>(
            bufU, hin, hidden + (long)l * B * H,
            sruB + (long)l * 2 * H, sruB + (long)l * 2 * H + H,
            hot, hidA + (long)l * B * H, T, B, H);
        float* tmp = hin; hin = hot; hot = tmp;
    }

    // [T,B,H] -> [B,T,H] into bufU region (U no longer needed)
    float* hbt = bufU;
    permute_swap01<<<(M * H + 255) / 256, blk, 0, stream>>>(hbt, hin, T, B, H);

    // W3: fx (pre-activation) to d_out, y = relu(fx) to free ws buffer
    float* y = hot;
    gemm_bf16x3<<<dim3((H + TILE_N - 1) / TILE_N, M / TILE_M), blk, 0, stream>>>(
        hbt, W3, b3, y, fx, M, H, H, 1);

    // W4: logits
    gemm_bf16x3<<<dim3((OUT + TILE_N - 1) / TILE_N, M / TILE_M), blk, 0, stream>>>(
        y, W4, b4, logits, nullptr, M, OUT, H, 0);
}
